// MoELayer_tp_65403761984106
// MI455X (gfx1250) — matrix operand order repaired and checked by reading, not compiled, not run on hardware
//
#include <hip/hip_runtime.h>
#include <hip/hip_bf16.h>
#include <math.h>

// ---------------------------------------------------------------- types
typedef __bf16 bf16;
typedef __bf16 v16bf __attribute__((ext_vector_type(16)));
typedef __bf16 v8bf  __attribute__((ext_vector_type(8)));
typedef float  v8f   __attribute__((ext_vector_type(8)));
typedef unsigned int uint32;
typedef uint32 v4u __attribute__((ext_vector_type(4)));
typedef int    v4i __attribute__((ext_vector_type(4)));
typedef int    v8i __attribute__((ext_vector_type(8)));

#define S_  2048
#define B_  4
#define H_  1024
#define NH_ 16
#define HD_ 64
#define E_  8
#define F_  4096
#define M_  (S_*B_)   // 8192 tokens

#if __has_builtin(__builtin_amdgcn_tensor_load_to_lds) && __has_builtin(__builtin_amdgcn_s_wait_tensorcnt)
#define USE_TDM 1
#else
#define USE_TDM 0
#endif

union Frag16 { v16bf v; v8bf h[2]; };

__device__ __forceinline__ v8f wmma_bf16(const Frag16& a, const Frag16& b, v8f c) {
  return __builtin_amdgcn_wmma_f32_16x16x32_bf16(false, a.v, false, b.v, (short)0, c, false, false);
}

// ---- 16-lane butterfly reductions: pure-VALU DPP (no LDS traffic)
#if __has_builtin(__builtin_amdgcn_update_dpp)
template<int CTRL>
__device__ __forceinline__ float dpp_mov_f(float x) {
  return __int_as_float(__builtin_amdgcn_update_dpp(0, __float_as_int(x), CTRL, 0xF, 0xF, true));
}
__device__ __forceinline__ float red16_max(float v) {
  v = fmaxf(v, dpp_mov_f<0xB1>(v));    // quad_perm(1,0,3,2)  : xor 1
  v = fmaxf(v, dpp_mov_f<0x4E>(v));    // quad_perm(2,3,0,1)  : xor 2
  v = fmaxf(v, dpp_mov_f<0x141>(v));   // row_half_mirror     : pairs across 4-groups
  v = fmaxf(v, dpp_mov_f<0x140>(v));   // row_mirror          : pairs across 8-halves
  return v;
}
__device__ __forceinline__ float red16_sum(float v) {
  v += dpp_mov_f<0xB1>(v);
  v += dpp_mov_f<0x4E>(v);
  v += dpp_mov_f<0x141>(v);
  v += dpp_mov_f<0x140>(v);
  return v;
}
#else
__device__ __forceinline__ float red16_max(float v) {
  #pragma unroll
  for (int off = 1; off < 16; off <<= 1) v = fmaxf(v, __shfl_xor(v, off, 32));
  return v;
}
__device__ __forceinline__ float red16_sum(float v) {
  #pragma unroll
  for (int off = 1; off < 16; off <<= 1) v += __shfl_xor(v, off, 32);
  return v;
}
#endif

// ---- Tensor Data Mover: DMA a [rows=128 x 32] bf16 tile (row stride K elems)
// into LDS with 16B pad after every 64B row (matches LD=40-element LDS stride).
#if USE_TDM
__device__ __forceinline__ void tdm_load_a_tile(
    const bf16* gsrc, int K, int Mtot, void* ldsDst)
{
  unsigned long long ga = (unsigned long long)(uintptr_t)gsrc;
  uint32 lds = (uint32)(uintptr_t)ldsDst;   // low 32 bits of flat LDS addr = LDS byte offset
  v4u g0;
  g0.x = 1u;                                             // count=1, user-mode descriptor
  g0.y = lds;                                            // lds_addr (bytes)
  g0.z = (uint32)(ga & 0xffffffffu);                     // global_addr[31:0]
  g0.w = (uint32)((ga >> 32) & 0x1ffffffu) | (2u << 30); // global_addr[56:32], type=2
  uint32 uk = (uint32)K, um = (uint32)Mtot;
  v8i g1;
  g1[0] = (int)((1u << 16)        // data_size = 2 bytes
              | (1u << 20)        // pad_enable
              | (3u << 22)        // pad_interval: 16 DWORDs (= 64B tile row)
              | (3u << 25));      // pad_amount: 4 DWORDs (= 16B -> 40-elem stride)
  g1[1] = (int)((uk & 0xffffu) << 16);                       // tensor_dim0[15:0]
  g1[2] = (int)((uk >> 16) | ((um & 0xffffu) << 16));        // tensor_dim0[31:16] | tensor_dim1[15:0]
  g1[3] = (int)((um >> 16) | (32u << 16));                   // tensor_dim1[31:16] | tile_dim0=32
  g1[4] = (int)(128u);                                       // tile_dim1=128
  g1[5] = (int)uk;                                           // tensor_dim0_stride[31:0]
  g1[6] = 0;                                                 // stride[47:32] | dim1_stride lo
  g1[7] = 0;
  v4i z4 = {0, 0, 0, 0};
  v8i z8 = {0, 0, 0, 0, 0, 0, 0, 0};
  // clang-23 / therock-10.0 arity: (g0, g1, g2, g3, g4, cpol)
  __builtin_amdgcn_tensor_load_to_lds(g0, g1, z4, z4, z8, 0);
}
#endif

// ---------------------------------------------------------------- RMSNorm -> bf16
__global__ __launch_bounds__(256) void rmsnorm_to_bf16(
    const float* __restrict__ x, const float* __restrict__ w, bf16* __restrict__ out)
{
  __shared__ float red[256];
  const int t = blockIdx.x, tid = threadIdx.x;
  const float* xp = x + (size_t)t * H_;
  float ss = 0.f;
  for (int i = tid; i < H_; i += 256) { float v = xp[i]; ss += v * v; }
  red[tid] = ss; __syncthreads();
  for (int s = 128; s > 0; s >>= 1) { if (tid < s) red[tid] += red[tid + s]; __syncthreads(); }
  float inv = rsqrtf(red[0] / (float)H_ + 1e-6f);
  for (int i = tid; i < H_; i += 256)
    out[(size_t)t * H_ + i] = (bf16)(xp[i] * inv * w[i]);
}

// ---------------------------------------------------------------- tiled WMMA GEMM
// C[M,N] = A[M,K](bf16, A-tile via TDM DMA) * B[K,N](fp32 -> bf16 into LDS)
// Double-buffered LDS, one barrier per BK step.
enum { EPI_BF16 = 0, EPI_RESID_DUP = 1, EPI_GELU = 2, EPI_GATED = 3 };

template<int EPI>
__global__ __launch_bounds__(256) void gemm_bf16_wmma(
    const bf16* __restrict__ A, const float* __restrict__ B,
    int M, int N, int K,
    float* __restrict__ outF, bf16* __restrict__ outB,
    const float* __restrict__ resid, float* __restrict__ dup,
    const float* __restrict__ gates, int gstride)
{
  constexpr int BM = 128, BN = 128, BK = 32, LD = BK + 8;
  __shared__ bf16 As[2][BM * LD];
  __shared__ bf16 Bs[2][BN * LD];   // stored transposed: [n][k]
  const int tid  = threadIdx.x;
  const int lane = tid & 31, wave = tid >> 5;
  const int wm = wave >> 1, wn = wave & 1;          // 4x2 wave grid
  const int hs = lane >> 4, l16 = lane & 15;
  const int rowA0 = blockIdx.y * BM, colB0 = blockIdx.x * BN;
  const int kTiles = K / BK;

  v8f acc[2][4] = {};
  float4 breg[4];
#if !USE_TDM
  v8bf areg[2];
#endif

  auto loadBregs = [&](int kt) {
    #pragma unroll
    for (int p = 0; p < 4; p++) {
      int lin = (p * 256 + tid) * 4;
      int r = lin >> 7, c = lin & 127;
      breg[p] = *(const float4*)(B + (size_t)(kt * BK + r) * N + colB0 + c);
    }
  };
  auto storeBlds = [&](int buf) {
    #pragma unroll
    for (int p = 0; p < 4; p++) {
      int lin = (p * 256 + tid) * 4;
      int r = lin >> 7, c = lin & 127;
      Bs[buf][(c + 0) * LD + r] = (bf16)breg[p].x;
      Bs[buf][(c + 1) * LD + r] = (bf16)breg[p].y;
      Bs[buf][(c + 2) * LD + r] = (bf16)breg[p].z;
      Bs[buf][(c + 3) * LD + r] = (bf16)breg[p].w;
    }
  };
#if !USE_TDM
  auto loadAregs = [&](int kt) {
    #pragma unroll
    for (int p = 0; p < 2; p++) {
      int lin = p * 256 + tid;
      int r = lin >> 2, ck = lin & 3;
      areg[p] = *(const v8bf*)(A + (size_t)(rowA0 + r) * K + kt * BK + ck * 8);
    }
  };
  auto storeAlds = [&](int buf) {
    #pragma unroll
    for (int p = 0; p < 2; p++) {
      int lin = p * 256 + tid;
      int r = lin >> 2, ck = lin & 3;
      *(v8bf*)(&As[buf][r * LD + ck * 8]) = areg[p];
    }
  };
#endif

  // ---- prologue: stage tile 0 into buffer 0
#if USE_TDM
  if (wave == 0)
    tdm_load_a_tile(A + (size_t)rowA0 * K, K, M, &As[0][0]);
#else
  loadAregs(0);
#endif
  loadBregs(0);
  storeBlds(0);
#if !USE_TDM
  storeAlds(0);
#endif
#if USE_TDM
  if (wave == 0) __builtin_amdgcn_s_wait_tensorcnt(0);
#endif
  __syncthreads();

  for (int kt = 0; kt < kTiles; kt++) {
    const int cur = kt & 1, nxt = cur ^ 1;
    // ---- start fetch of tile kt+1 (overlaps with WMMA below)
    if (kt + 1 < kTiles) {
#if USE_TDM
      if (wave == 0)
        tdm_load_a_tile(A + (size_t)rowA0 * K + (size_t)(kt + 1) * BK, K, M, &As[nxt][0]);
#else
      loadAregs(kt + 1);
#endif
      loadBregs(kt + 1);
    }
    if (kt + 2 < kTiles) {   // L2 prefetch of the B tile after next
      int r = tid >> 3, c = (tid & 7) * 16;
      __builtin_prefetch(B + (size_t)((kt + 2) * BK + r) * N + colB0 + c, 0, 1);
    }

    // ---- compute on current buffer
    Frag16 af[2], bfr[4];
    #pragma unroll
    for (int tm = 0; tm < 2; tm++) {
      const bf16* p = &As[cur][(wm * 32 + tm * 16 + l16) * LD + hs * 8];
      af[tm].h[0] = *(const v8bf*)p;           // K = hs*8 .. +7
      af[tm].h[1] = *(const v8bf*)(p + 16);    // K = 16+hs*8 .. +7
    }
    #pragma unroll
    for (int tn = 0; tn < 4; tn++) {
      const bf16* p = &Bs[cur][(wn * 64 + tn * 16 + l16) * LD + hs * 8];
      bfr[tn].h[0] = *(const v8bf*)p;          // K = hs*8 .. +7
      bfr[tn].h[1] = *(const v8bf*)(p + 16);   // K = 16+hs*8 .. +7
    }
    #pragma unroll
    for (int tm = 0; tm < 2; tm++)
      #pragma unroll
      for (int tn = 0; tn < 4; tn++)
        acc[tm][tn] = wmma_bf16(af[tm], bfr[tn], acc[tm][tn]);

    // ---- commit tile kt+1 into the other buffer, single barrier per step
    if (kt + 1 < kTiles) {
      storeBlds(nxt);
#if !USE_TDM
      storeAlds(nxt);
#endif
    }
#if USE_TDM
    if (wave == 0) __builtin_amdgcn_s_wait_tensorcnt(0);
#endif
    __syncthreads();
  }

  // ---- epilogue (C layout: row = r + 8*hs, col = l16 per 16x16 tile)
  #pragma unroll
  for (int tm = 0; tm < 2; tm++)
    #pragma unroll
    for (int r = 0; r < 8; r++) {
      int row = rowA0 + wm * 32 + tm * 16 + r + hs * 8;
      float gate = 0.f;
      if (EPI == EPI_GATED) gate = gates[(size_t)row * gstride];
      #pragma unroll
      for (int tn = 0; tn < 4; tn++) {
        int col = colB0 + wn * 64 + tn * 16 + l16;
        size_t o = (size_t)row * N + col;
        float va = acc[tm][tn][r];
        if (EPI == EPI_BF16) {
          outB[o] = (bf16)va;
        } else if (EPI == EPI_RESID_DUP) {
          float x = va + resid[o];
          outF[o] = x; dup[o] = x;
        } else if (EPI == EPI_GELU) {
          float g = 0.5f * va * (1.0f + erff(va * 0.70710678118f));
          outB[o] = (bf16)g;
        } else { // EPI_GATED
          outF[o] += gate * va;
        }
      }
    }
}

// ---------------------------------------------------------------- RoPE + reorder
// qkv bf16 [S,B,3,NH,HD] -> Q,K bf16 [B,NH,S,HD], V^T bf16 [B,NH,HD,S]
__global__ __launch_bounds__(256) void rope_reorder(
    const bf16* __restrict__ qkv, bf16* __restrict__ Q,
    bf16* __restrict__ Kk, bf16* __restrict__ Vt)
{
  int idx = blockIdx.x * 256 + threadIdx.x;
  int d = idx & (HD_ - 1);
  int h = (idx >> 6) & (NH_ - 1);
  int b = (idx >> 10) & (B_ - 1);
  int s = idx >> 12;
  const bf16* base = qkv + (size_t)(s * B_ + b) * 3 * H_;
  int hd = h * HD_;
  float q = (float)base[hd + d];
  float k = (float)base[H_ + hd + d];
  float v = (float)base[2 * H_ + hd + d];
  int dro = (d < 32) ? d + 32 : d - 32;
  float qr = (float)base[hd + dro];        if (d < 32) qr = -qr;
  float kr = (float)base[H_ + hd + dro];   if (d < 32) kr = -kr;
  int di = d & 31;
  float invf = powf(10000.0f, -(float)(2 * di) / (float)HD_);
  float fr = (float)s * invf;
  float c = cosf(fr), sn = sinf(fr);
  size_t bh = (size_t)(b * NH_ + h);
  size_t o = (bh * S_ + s) * HD_ + d;
  Q[o]  = (bf16)(q * c + qr * sn);
  Kk[o] = (bf16)(k * c + kr * sn);
  Vt[(bh * HD_ + d) * S_ + s] = (bf16)v;
}

// ---------------------------------------------------------------- flash attention
// one block per (qtile of 128, head, batch); each wave owns 16 q-rows independently
__global__ __launch_bounds__(256) void flash_attn(
    const bf16* __restrict__ Q, const bf16* __restrict__ Kt,
    const bf16* __restrict__ Vt, bf16* __restrict__ ctx)
{
  __shared__ bf16 Pbuf[8][16 * 40];
  const int tid = threadIdx.x, lane = tid & 31, wave = tid >> 5;
  const int hs = lane >> 4, l16 = lane & 15;
  const int h = blockIdx.y, b = blockIdx.z;
  const int q0 = blockIdx.x * 128 + wave * 16;
  const size_t bh = (size_t)(b * NH_ + h);
  const bf16* Qp = Q  + bh * S_ * HD_;
  const bf16* Kp = Kt + bh * S_ * HD_;
  const bf16* Vp = Vt + bh * HD_ * S_;

  // Q A-fragments (held in registers for the whole kernel)
  Frag16 aq[2];
  {
    const bf16* p = Qp + (size_t)(q0 + l16) * HD_;
    #pragma unroll
    for (int kc = 0; kc < 2; kc++) {
      aq[kc].h[0] = *(const v8bf*)(p + kc * 32 + hs * 8);
      aq[kc].h[1] = *(const v8bf*)(p + kc * 32 + hs * 8 + 16);
    }
  }

  float mrow[8], lrow[8];
  #pragma unroll
  for (int r = 0; r < 8; r++) { mrow[r] = -1e30f; lrow[r] = 0.f; }
  v8f acco[4] = {};

  const int nkb = (q0 + 16 + 31) >> 5;           // wave-uniform trip count
  for (int jb = 0; jb < nkb; jb++) {
    int kbase = jb * 32;
    // ---- scores: 16x32 via two 16x16 WMMA C-tiles, K = HD = 64 (2 chunks)
    v8f sc[2] = {};
    #pragma unroll
    for (int kt = 0; kt < 2; kt++) {
      const bf16* p = Kp + (size_t)(kbase + kt * 16 + l16) * HD_;
      #pragma unroll
      for (int kc = 0; kc < 2; kc++) {
        Frag16 bk;
        bk.h[0] = *(const v8bf*)(p + kc * 32 + hs * 8);
        bk.h[1] = *(const v8bf*)(p + kc * 32 + hs * 8 + 16);
        sc[kt] = wmma_bf16(aq[kc], bk, sc[kt]);
      }
    }
    // ---- online softmax (fp32), DPP butterfly reduces each 16-lane row
    float pv[2][8];
    #pragma unroll
    for (int r = 0; r < 8; r++) {
      int row = q0 + r + hs * 8;
      float s0 = sc[0][r] * 0.125f; if (kbase + l16      > row) s0 = -1e30f;
      float s1 = sc[1][r] * 0.125f; if (kbase + 16 + l16 > row) s1 = -1e30f;
      float mx = red16_max(fmaxf(s0, s1));
      float mnew = fmaxf(mrow[r], mx);
      float scl = __expf(mrow[r] - mnew);
      mrow[r] = mnew;
      float p0 = __expf(s0 - mnew), p1 = __expf(s1 - mnew);
      float rs = red16_sum(p0 + p1);
      lrow[r] = lrow[r] * scl + rs;
      #pragma unroll
      for (int tn = 0; tn < 4; tn++) acco[tn][r] = acco[tn][r] * scl;
      pv[0][r] = p0; pv[1][r] = p1;
    }
    // ---- P: C-layout -> A-fragment via per-wave LDS round trip (intra-wave only)
    bf16* pb = Pbuf[wave];
    #pragma unroll
    for (int kt = 0; kt < 2; kt++)
      #pragma unroll
      for (int r = 0; r < 8; r++)
        pb[(r + hs * 8) * 40 + kt * 16 + l16] = (bf16)pv[kt][r];
#if __has_builtin(__builtin_amdgcn_s_wait_dscnt)
    __builtin_amdgcn_s_wait_dscnt(0);
#else
    asm volatile("s_wait_dscnt 0" ::: "memory");
#endif
    Frag16 ap;
    {
      const bf16* p = pb + l16 * 40 + hs * 8;
      ap.h[0] = *(const v8bf*)p;
      ap.h[1] = *(const v8bf*)(p + 16);
    }
    // ---- ctx += P @ V  (V^T layout gives contiguous per-lane B-fragments)
    #pragma unroll
    for (int tn = 0; tn < 4; tn++) {
      Frag16 bv;
      const bf16* p = Vp + (size_t)(tn * 16 + l16) * S_ + kbase + hs * 8;
      bv.h[0] = *(const v8bf*)p;
      bv.h[1] = *(const v8bf*)(p + 16);
      acco[tn] = wmma_bf16(ap, bv, acco[tn]);
    }
  }
  // ---- normalize, write ctx [S,B,H] bf16
  #pragma unroll
  for (int r = 0; r < 8; r++) {
    int srow = q0 + r + hs * 8;
    float invl = 1.f / lrow[r];
    size_t obase = ((size_t)srow * B_ + b) * H_ + h * HD_;
    #pragma unroll
    for (int tn = 0; tn < 4; tn++)
      ctx[obase + tn * 16 + l16] = (bf16)(acco[tn][r] * invl);
  }
}

// ---------------------------------------------------------------- router (RMSNorm2 + top2 gates)
__global__ __launch_bounds__(256) void router_gates(
    const float* __restrict__ xin, const float* __restrict__ w,
    const float* __restrict__ rw, bf16* __restrict__ x2b, float* __restrict__ gates)
{
  __shared__ float red[256];
  __shared__ float lg[E_];
  const int t = blockIdx.x, tid = threadIdx.x;
  const float* xp = xin + (size_t)t * H_;
  float ss = 0.f;
  for (int i = tid; i < H_; i += 256) { float v = xp[i]; ss += v * v; }
  red[tid] = ss; __syncthreads();
  for (int s = 128; s > 0; s >>= 1) { if (tid < s) red[tid] += red[tid + s]; __syncthreads(); }
  float inv = rsqrtf(red[0] / (float)H_ + 1e-6f);
  __syncthreads();
  float pl[E_];
  #pragma unroll
  for (int e = 0; e < E_; e++) pl[e] = 0.f;
  for (int i = tid; i < H_; i += 256) {
    float xn = xp[i] * inv * w[i];
    x2b[(size_t)t * H_ + i] = (bf16)xn;
    #pragma unroll
    for (int e = 0; e < E_; e++) pl[e] += xn * rw[i * E_ + e];
  }
  for (int e = 0; e < E_; e++) {
    red[tid] = pl[e]; __syncthreads();
    for (int s = 128; s > 0; s >>= 1) { if (tid < s) red[tid] += red[tid + s]; __syncthreads(); }
    if (tid == 0) lg[e] = red[0];
    __syncthreads();
  }
  if (tid == 0) {
    float mx = lg[0];
    for (int e = 1; e < E_; e++) mx = fmaxf(mx, lg[e]);
    float pr[E_], sum = 0.f;
    for (int e = 0; e < E_; e++) { pr[e] = __expf(lg[e] - mx); sum += pr[e]; }
    float isum = 1.f / sum;
    int i1 = 0; for (int e = 1; e < E_; e++) if (pr[e] > pr[i1]) i1 = e;
    int i2 = (i1 == 0) ? 1 : 0;
    for (int e = 0; e < E_; e++) if (e != i1 && pr[e] > pr[i2]) i2 = e;
    for (int e = 0; e < E_; e++)
      gates[(size_t)t * E_ + e] = (e == i1 || e == i2) ? pr[e] * isum : 0.f;
  }
}

// ---------------------------------------------------------------- launcher
extern "C" void kernel_launch(void* const* d_in, const int* in_sizes, int n_in,
                              void* d_out, int out_size, void* d_ws, size_t ws_size,
                              hipStream_t stream)
{
  (void)in_sizes; (void)n_in; (void)out_size; (void)ws_size;
  const float* hidden = (const float*)d_in[0];
  const float* ln1    = (const float*)d_in[1];
  const float* ln2    = (const float*)d_in[2];
  const float* Wqkv   = (const float*)d_in[3];
  const float* Wo     = (const float*)d_in[4];
  const float* rw     = (const float*)d_in[5];
  const float* W1     = (const float*)d_in[6];
  const float* W2     = (const float*)d_in[7];
  float* out = (float*)d_out;

  char* base = (char*)d_ws;
  size_t off = 0;
  auto alloc = [&](size_t bytes) -> char* {
    char* p = base + off; off = (off + bytes + 255) & ~(size_t)255; return p;
  };
  bf16*  xb       = (bf16*) alloc((size_t)M_ * H_ * 2);
  bf16*  qkvb     = (bf16*) alloc((size_t)M_ * 3 * H_ * 2);
  bf16*  Qb       = (bf16*) alloc((size_t)M_ * H_ * 2);
  bf16*  Kb       = (bf16*) alloc((size_t)M_ * H_ * 2);
  bf16*  Vtb      = (bf16*) alloc((size_t)M_ * H_ * 2);
  bf16*  ctxb     = (bf16*) alloc((size_t)M_ * H_ * 2);
  float* attn_out = (float*)alloc((size_t)M_ * H_ * 4);
  bf16*  x2b      = (bf16*) alloc((size_t)M_ * H_ * 2);
  float* gates    = (float*)alloc((size_t)M_ * E_ * 4);
  bf16*  hb       = (bf16*) alloc((size_t)M_ * F_ * 2);

  rmsnorm_to_bf16<<<M_, 256, 0, stream>>>(hidden, ln1, xb);

  gemm_bf16_wmma<EPI_BF16><<<dim3(3 * H_ / 128, M_ / 128), 256, 0, stream>>>(
      xb, Wqkv, M_, 3 * H_, H_, nullptr, qkvb, nullptr, nullptr, nullptr, 0);

  rope_reorder<<<(S_ * B_ * NH_ * HD_) / 256, 256, 0, stream>>>(qkvb, Qb, Kb, Vtb);

  flash_attn<<<dim3(S_ / 128, NH_, B_), 256, 0, stream>>>(Qb, Kb, Vtb, ctxb);

  gemm_bf16_wmma<EPI_RESID_DUP><<<dim3(H_ / 128, M_ / 128), 256, 0, stream>>>(
      ctxb, Wo, M_, H_, H_, attn_out, nullptr, hidden, out, nullptr, 0);

  router_gates<<<M_, 256, 0, stream>>>(attn_out, ln2, rw, x2b, gates);

  for (int e = 0; e < E_; e++) {
    gemm_bf16_wmma<EPI_GELU><<<dim3(F_ / 128, M_ / 128), 256, 0, stream>>>(
        x2b, W1 + (size_t)e * H_ * F_, M_, F_, H_, nullptr, hb, nullptr, nullptr, nullptr, 0);
    gemm_bf16_wmma<EPI_GATED><<<dim3(H_ / 128, M_ / 128), 256, 0, stream>>>(
        hb, W2 + (size_t)e * F_ * H_, M_, H_, F_, out, nullptr, nullptr, nullptr, gates + e, E_);
  }
}
